// EGRRBlock_31988916420666
// MI455X (gfx1250) — matrix operand order repaired and checked by reading, not compiled, not run on hardware
//
#include <hip/hip_runtime.h>
#include <hip/hip_bf16.h>
#include <math.h>

// ---------------------------------------------------------------------------
// EGRR block for MI455X (gfx1250, wave32).
// Channel-mix 1x1 conv is done with v_wmma_f32_16x16x32_f16 (the codegen-
// confirmed CDNA5 matrix path).  All reductions are deterministic (no float
// atomics).  Whole working set (h, z, z2 ~ 154MB) fits in the 192MB L2, so
// the multi-kernel pipeline stays cache-resident between stages.
// ---------------------------------------------------------------------------

typedef __attribute__((ext_vector_type(16))) _Float16 v16h;
typedef __attribute__((ext_vector_type(8)))  _Float16 v8h;
typedef __attribute__((ext_vector_type(8)))  float    v8f;

#define NB    8
#define CCH   128
#define HH    112
#define WW    112
#define HWSZ  (HH * WW)            // 12544
#define PTILE 64                   // spatial positions per GEMM workgroup
#define PH    37                   // 112/3 (VALID pool)
#define NPOOL (PH * PH)            // 1369

// ---------------------------------------------------------------------------
// Prep: symmetric f16 weight  Wsym[d][c] = 0.5*(W[d][c] + W[c][d])
// ---------------------------------------------------------------------------
__global__ __launch_bounds__(256) void wsym_kernel(const float* __restrict__ w,
                                                   _Float16* __restrict__ wsym) {
    int i = blockIdx.x * 256 + threadIdx.x;          // 0..16383
    int d = i >> 7, c = i & 127;
    wsym[i] = (_Float16)(0.5f * (w[d * CCH + c] + w[c * CCH + d]));
}

// ---------------------------------------------------------------------------
// GAP: gap[n*C+c] = mean over HxW of h   (deterministic tree reduction)
// ---------------------------------------------------------------------------
__global__ __launch_bounds__(256) void gap_kernel(const float* __restrict__ h,
                                                  float* __restrict__ gap) {
    int nc = blockIdx.x;                             // n*C + c  (1024 blocks)
    const float* base = h + (size_t)nc * HWSZ;
    float s = 0.f;
    for (int i = threadIdx.x; i < HWSZ; i += 256) s += base[i];
    __shared__ float red[256];
    red[threadIdx.x] = s;
    __syncthreads();
    for (int st = 128; st > 0; st >>= 1) {
        if ((int)threadIdx.x < st) red[threadIdx.x] += red[threadIdx.x + st];
        __syncthreads();
    }
    if (threadIdx.x == 0) gap[nc] = red[0] * (1.0f / HWSZ);
}

// ---------------------------------------------------------------------------
// Entropy: one wave32 per pooled position; 4 channels per lane; cross-lane
// log-softmax over C=128; write -sum(p*logp) per position (no atomics).
// ---------------------------------------------------------------------------
__global__ __launch_bounds__(256) void ent_kernel(const float* __restrict__ h,
                                                  float* __restrict__ entbuf) {
    int wave = threadIdx.x >> 5, lane = threadIdx.x & 31;
    int gw = blockIdx.x * 8 + wave;                  // 1369*8 = 10952 exactly
    if (gw >= NB * NPOOL) return;
    int n = gw / NPOOL, rem = gw % NPOOL;
    int ph = rem / PH, pw = rem % PH;

    float pv[4];
    for (int k = 0; k < 4; ++k) {
        int c = lane * 4 + k;
        const float* hp = h + ((size_t)n * CCH + c) * HWSZ + (ph * 3) * WW + pw * 3;
        float s = 0.f;
        for (int dy = 0; dy < 3; ++dy)
            for (int dx = 0; dx < 3; ++dx) s += hp[dy * WW + dx];
        pv[k] = s * (1.0f / 9.0f);
    }
    float m = fmaxf(fmaxf(pv[0], pv[1]), fmaxf(pv[2], pv[3]));
    for (int off = 16; off > 0; off >>= 1) m = fmaxf(m, __shfl_xor(m, off));
    float se = 0.f;
    for (int k = 0; k < 4; ++k) se += expf(pv[k] - m);
    for (int off = 16; off > 0; off >>= 1) se += __shfl_xor(se, off);
    float logZ = m + logf(se);
    float part = 0.f;
    for (int k = 0; k < 4; ++k) { float lp = pv[k] - logZ; part += expf(lp) * lp; }
    for (int off = 16; off > 0; off >>= 1) part += __shfl_xor(part, off);
    if (lane == 0) entbuf[gw] = -part;
}

// ---------------------------------------------------------------------------
// Gates: logits = gap @ gate_W + gate_b + mean(ent)*gate_we ; softmax(3)
// ---------------------------------------------------------------------------
__global__ void gates_kernel(const float* __restrict__ gap,
                             const float* __restrict__ entbuf,
                             const float* __restrict__ gW,
                             const float* __restrict__ gb,
                             const float* __restrict__ gwe,
                             float* __restrict__ gates) {
    int n = threadIdx.x;
    if (n >= NB) return;
    float ent = 0.f;
    for (int i = 0; i < NPOOL; ++i) ent += entbuf[n * NPOOL + i];
    ent *= (1.0f / (float)NPOOL);
    float l[3];
    for (int j = 0; j < 3; ++j) {
        float s = gb[j] + ent * gwe[j];
        for (int c = 0; c < CCH; ++c) s += gap[n * CCH + c] * gW[c * 3 + j];
        l[j] = s;
    }
    float m = fmaxf(l[0], fmaxf(l[1], l[2]));
    float e0 = expf(l[0] - m), e1 = expf(l[1] - m), e2 = expf(l[2] - m);
    float inv = 1.0f / (e0 + e1 + e2);
    gates[n * 3 + 0] = e0 * inv;
    gates[n * 3 + 1] = e1 * inv;
    gates[n * 3 + 2] = e2 * inv;
}

// ---------------------------------------------------------------------------
// Channel-mix GEMM via WMMA:  z[n,d,p] = sum_c Wsym[d,c]*h[n,c,p] + bias[d]
// One WG = 256 threads (8 waves) -> 128 out-channels x 64 positions.
// A = Wsym rows (16-bit A 16x32 layout: lane M = lane&15; halves 0..7 hold
//     K = k0 + (lane<16?0:8)..+7, halves 8..15 hold K = k0+16+(lane<16?0:8)..+7).
// B from LDS tile lds[pos][chan]: lane N = lane&15; same K layout as A:
//     halves 0..7 K = k0+(lane<16?0:8)..+7, halves 8..15 K = k0+16+(lane<16?0:8)..+7.
// ---------------------------------------------------------------------------
__global__ __launch_bounds__(256) void gemm_kernel(const float* __restrict__ h,
                                                   const _Float16* __restrict__ wsym,
                                                   const float* __restrict__ sym_bias,
                                                   float* __restrict__ z) {
    __shared__ _Float16 lds[PTILE * CCH];            // [p][c]  16 KB
    const int tid  = threadIdx.x;
    const int tile = blockIdx.x;                     // 8 * 196 tiles
    const int n  = tile / (HWSZ / PTILE);
    const int p0 = (tile % (HWSZ / PTILE)) * PTILE;
    const float* hn = h + (size_t)n * CCH * HWSZ;

    // Stage h tile -> LDS (f32 -> f16), coalesced over positions.
    for (int i = 0; i < (PTILE * CCH) / 256; ++i) {
        int flat = i * 256 + tid;
        int c = flat >> 6;                           // /64
        int p = flat & 63;
        lds[p * CCH + c] = (_Float16)hn[(size_t)c * HWSZ + p0 + p];
    }
    __syncthreads();

    const int wave = tid >> 5;
    const int lane = tid & 31;
    const int nrow = lane & 15;                      // M for A, N for B/D
    const int hi   = lane >> 4;                      // lane group 0/1
    const int dbase = wave * 16;

    // Preload the 4 K-step A fragments (Wsym row for this lane's M).
    v16h afrag[4];
    const _Float16* wrow = wsym + (dbase + nrow) * CCH;
    for (int k = 0; k < 4; ++k) {
        int c0 = k * 32 + hi * 8;
        v8h lo = *(const v8h*)(wrow + c0);
        v8h hh = *(const v8h*)(wrow + c0 + 16);
        afrag[k] = __builtin_shufflevector(lo, hh, 0, 1, 2, 3, 4, 5, 6, 7,
                                                   8, 9, 10, 11, 12, 13, 14, 15);
    }

    v8f acc[4] = {};                                 // 4 N-subtiles of 16 pos
    for (int k = 0; k < 4; ++k) {
        int cb = k * 32 + hi * 8;
        for (int j = 0; j < 4; ++j) {
            int pl = j * 16 + nrow;
            const _Float16* src = &lds[pl * CCH + cb];
            v8h b0 = *(const v8h*)(src);
            v8h b1 = *(const v8h*)(src + 16);
            v16h b = __builtin_shufflevector(b0, b1, 0, 1, 2, 3, 4, 5, 6, 7,
                                                     8, 9, 10, 11, 12, 13, 14, 15);
            acc[j] = __builtin_amdgcn_wmma_f32_16x16x32_f16(
                false, afrag[k], false, b, (short)0, acc[j], false, false);
        }
    }

    // Store: D layout -> N = lane&15 (position), M = r + hi*8 (channel).
    float* zn = z + (size_t)n * CCH * HWSZ;
    for (int j = 0; j < 4; ++j) {
        int pl = j * 16 + nrow;
        for (int r = 0; r < 8; ++r) {
            int dd = dbase + hi * 8 + r;
            zn[(size_t)dd * HWSZ + p0 + pl] = acc[j][r] + sym_bias[dd];
        }
    }
}

// ---------------------------------------------------------------------------
// Gate-weighted shared depthwise conv (dilations 1,2,4) + per-row IN stats.
// One block per (n,c,row); 128 threads (112 active columns). No atomics.
// ---------------------------------------------------------------------------
__global__ __launch_bounds__(128) void dwconv_kernel(const float* __restrict__ z,
                                                     const float* __restrict__ dw_w,
                                                     const float* __restrict__ dw_b,
                                                     const float* __restrict__ gates,
                                                     float* __restrict__ z2,
                                                     float* __restrict__ rowstats) {
    int b  = blockIdx.x;                             // (n*C + c)*H + y
    int y  = b % HH;
    int nc = b / HH;
    int c  = nc % CCH;
    int n  = nc / CCH;
    int x  = threadIdx.x;

    float val = 0.f;
    if (x < WW) {
        const float* zp = z + (size_t)nc * HWSZ;
        float w[9];
        for (int i = 0; i < 9; ++i) w[i] = dw_w[c * 9 + i];
        float g0 = gates[n * 3 + 0], g1 = gates[n * 3 + 1], g2 = gates[n * 3 + 2];
        float bias = dw_b[c];
        const int dil[3] = {1, 2, 4};
        float conv[3];
        for (int di = 0; di < 3; ++di) {
            int dd = dil[di];
            float acc = 0.f;
            for (int ky = 0; ky < 3; ++ky)
                for (int kx = 0; kx < 3; ++kx) {
                    int yy = y + (ky - 1) * dd, xx = x + (kx - 1) * dd;
                    if (yy >= 0 && yy < HH && xx >= 0 && xx < WW)
                        acc += w[ky * 3 + kx] * zp[yy * WW + xx];
                }
            conv[di] = acc;
        }
        val = g0 * (conv[0] + bias) + g1 * (conv[1] + bias) + g2 * (conv[2] + bias);
        z2[(size_t)nc * HWSZ + y * WW + x] = val;
    }

    __shared__ float rs[128], rq[128];
    rs[x] = (x < WW) ? val : 0.f;
    rq[x] = (x < WW) ? val * val : 0.f;
    __syncthreads();
    for (int st = 64; st > 0; st >>= 1) {
        if (x < st) { rs[x] += rs[x + st]; rq[x] += rq[x + st]; }
        __syncthreads();
    }
    if (x == 0) {
        rowstats[2 * (nc * HH + y) + 0] = rs[0];
        rowstats[2 * (nc * HH + y) + 1] = rq[0];
    }
}

// Reduce row stats -> per-(n,c) sum / sumsq (deterministic).
__global__ __launch_bounds__(128) void redstats_kernel(const float* __restrict__ rowstats,
                                                       float* __restrict__ stats) {
    int nc = blockIdx.x;                             // 1024 blocks
    int t  = threadIdx.x;
    float s = 0.f, q = 0.f;
    if (t < HH) {
        s = rowstats[2 * (nc * HH + t) + 0];
        q = rowstats[2 * (nc * HH + t) + 1];
    }
    __shared__ float rs[128], rq[128];
    rs[t] = s; rq[t] = q;
    __syncthreads();
    for (int st = 64; st > 0; st >>= 1) {
        if (t < st) { rs[t] += rs[t + st]; rq[t] += rq[t + st]; }
        __syncthreads();
    }
    if (t == 0) { stats[2 * nc] = rs[0]; stats[2 * nc + 1] = rq[0]; }
}

// ---------------------------------------------------------------------------
// Instance norm + ReLU6 + residual:  h += clip(norm(z2)*gamma + beta, 0, 6)
// ---------------------------------------------------------------------------
__global__ __launch_bounds__(256) void norm_kernel(const float* __restrict__ z2,
                                                   const float* __restrict__ stats,
                                                   const float* __restrict__ gamma,
                                                   const float* __restrict__ beta,
                                                   float* __restrict__ h, int t) {
    size_t idx = (size_t)blockIdx.x * 256 + threadIdx.x;   // exact fit
    size_t nc = idx / HWSZ;
    int c = (int)(nc % CCH);
    float s = stats[2 * nc], q = stats[2 * nc + 1];
    float mean = s * (1.0f / HWSZ);
    float var  = q * (1.0f / HWSZ) - mean * mean;
    float rsq  = rsqrtf(var + 1e-5f);
    float v = (z2[idx] - mean) * rsq * gamma[t * CCH + c] + beta[t * CCH + c];
    v = fminf(fmaxf(v, 0.0f), 6.0f);
    h[idx] += v;
}

// ---------------------------------------------------------------------------
extern "C" void kernel_launch(void* const* d_in, const int* in_sizes, int n_in,
                              void* d_out, int out_size, void* d_ws, size_t ws_size,
                              hipStream_t stream) {
    (void)in_sizes; (void)n_in; (void)out_size; (void)ws_size;

    const float* x     = (const float*)d_in[0];
    const float* dw_w  = (const float*)d_in[1];
    const float* dw_b  = (const float*)d_in[2];
    const float* sym_w = (const float*)d_in[3];
    const float* sym_b = (const float*)d_in[4];
    const float* gW    = (const float*)d_in[5];
    const float* gb    = (const float*)d_in[6];
    const float* gwe   = (const float*)d_in[7];
    const float* gamma = (const float*)d_in[8];
    const float* beta  = (const float*)d_in[9];

    float* h = (float*)d_out;
    const size_t NCHW = (size_t)NB * CCH * HWSZ;     // 12,845,056

    char* ws = (char*)d_ws;
    float*    z        = (float*)ws;       ws += NCHW * sizeof(float);
    float*    z2       = (float*)ws;       ws += NCHW * sizeof(float);
    _Float16* wsym     = (_Float16*)ws;    ws += (size_t)CCH * CCH * sizeof(_Float16);
    float*    gap      = (float*)ws;       ws += (size_t)NB * CCH * sizeof(float);
    float*    entbuf   = (float*)ws;       ws += (size_t)NB * NPOOL * sizeof(float);
    float*    gates    = (float*)ws;       ws += (size_t)NB * 3 * sizeof(float);
    float*    rowstats = (float*)ws;       ws += (size_t)NB * CCH * HH * 2 * sizeof(float);
    float*    stats    = (float*)ws;       ws += (size_t)NB * CCH * 2 * sizeof(float);

    // h <- x
    hipMemcpyAsync(h, x, NCHW * sizeof(float), hipMemcpyDeviceToDevice, stream);
    // Wsym in f16 (once).
    wsym_kernel<<<(CCH * CCH) / 256, 256, 0, stream>>>(sym_w, wsym);

    const int gemm_blocks = NB * (HWSZ / PTILE);     // 1568
    const int ent_blocks  = (NB * NPOOL) / 8;        // 1369 (exact)

    for (int t = 0; t < 4; ++t) {
        gap_kernel   <<<NB * CCH, 256, 0, stream>>>(h, gap);
        ent_kernel   <<<ent_blocks, 256, 0, stream>>>(h, entbuf);
        gates_kernel <<<1, 32, 0, stream>>>(gap, entbuf, gW, gb, gwe, gates);
        gemm_kernel  <<<gemm_blocks, 256, 0, stream>>>(h, wsym, sym_b, z);
        dwconv_kernel<<<NB * CCH * HH, 128, 0, stream>>>(z, dw_w, dw_b, gates, z2, rowstats);
        redstats_kernel<<<NB * CCH, 128, 0, stream>>>(rowstats, stats);
        norm_kernel  <<<(int)(NCHW / 256), 256, 0, stream>>>(z2, stats, gamma, beta, h, t);
    }
}
